// BroadcastZInvariantPointAttention_60146722013709
// MI455X (gfx1250) — matrix operand order repaired and checked by reading, not compiled, not run on hardware
//
#include <hip/hip_runtime.h>
#include <hip/hip_bf16.h>
#include <math.h>

#define NRES 768
#define CS   384
#define CZ   128
#define NH   12
#define CH   16
#define PQK  4
#define PV   8
#define CZ4  32

typedef __bf16 bf16;
typedef __attribute__((ext_vector_type(8)))  __bf16   v8bf;
typedef __attribute__((ext_vector_type(16))) __bf16   v16bf;
typedef __attribute__((ext_vector_type(16))) _Float16 v16h;
typedef __attribute__((ext_vector_type(8)))  float    v8f;
typedef __attribute__((ext_vector_type(4)))  float    v4f;

// ---- helpers -------------------------------------------------------------

__device__ inline v8f vzero8() {
  v8f a;
#pragma unroll
  for (int t = 0; t < 8; ++t) a[t] = 0.f;
  return a;
}

// A fragment (16-bit, 16x32): per-lane 16 contiguous K values
// caller passes  rowptr + k0 + (lane>>4)*16
__device__ inline v16bf a_frag_bf(const bf16* p) {
  v8bf lo = *(const v8bf*)p;
  v8bf hi = *(const v8bf*)(p + 8);
  v16bf r;
#pragma unroll
  for (int t = 0; t < 8; ++t) { r[t] = lo[t]; r[t + 8] = hi[t]; }
  return r;
}
__device__ inline v16h a_frag_h(const _Float16* p) {
  v16h r;
#pragma unroll
  for (int t = 0; t < 8; ++t) { r[t] = p[t]; r[t + 8] = p[8 + t]; }
  return r;
}
// B fragment (16-bit, 32x16) from [n][k]-major storage:
// caller passes  colptr + k0 + (lane>>4)*16  (16 contiguous K values)
__device__ inline v16bf b_frag_bf(const bf16* p) { return *(const v16bf*)p; }
__device__ inline v16h  b_frag_h(const _Float16* p) { return *(const v16h*)p; }

__device__ inline v8f wmma_bf(v16bf a, v16bf b, v8f c) {
  return __builtin_amdgcn_wmma_f32_16x16x32_bf16(false, a, false, b, (short)0, c, false, false);
}
__device__ inline v8f wmma_h(v16h a, v16h b, v8f c) {
  return __builtin_amdgcn_wmma_f32_16x16x32_f16(false, a, false, b, (short)0, c, false, false);
}

// ---- kernel 1: pack weights into bf16 [out][k] (transposed) --------------

__global__ void k_pack(const float* __restrict__ Wq, const float* __restrict__ Wk,
                       const float* __restrict__ Wv, const float* __restrict__ Wqp,
                       const float* __restrict__ Wkp, const float* __restrict__ Wvp,
                       const float* __restrict__ Wb, const float* __restrict__ Wdz,
                       const float* __restrict__ Wout, const float* __restrict__ Woutb,
                       const float* __restrict__ Woutpts,
                       bf16* __restrict__ WallT, bf16* __restrict__ WzT,
                       bf16* __restrict__ WfT) {
  int t = blockIdx.x * blockDim.x + threadIdx.x;
  if (t < 1152 * 384) {                       // WallT[o][k]: all s-projections
    int o = t / 384, k = t % 384;
    float v;
    if (o < 192)      v = Wq[k * 192 + o];
    else if (o < 384) v = Wk[k * 192 + (o - 192)];
    else if (o < 576) v = Wv[k * 192 + (o - 384)];
    else if (o < 720) v = Wqp[k * 144 + (o - 576)];
    else if (o < 864) v = Wkp[k * 144 + (o - 720)];
    else              v = Wvp[k * 288 + (o - 864)];
    WallT[o * 384 + k] = (bf16)v;
    return;
  }
  t -= 1152 * 384;
  if (t < 48 * 128) {                         // WzT[o][k]: [Wb|Wdz|pad]
    int o = t / 128, k = t % 128;
    float v = 0.f;
    if (o < 12)      v = Wb[k * 12 + o];
    else if (o < 44) v = Wdz[k * 32 + (o - 12)];
    WzT[o * 128 + k] = (bf16)v;
    return;
  }
  t -= 48 * 128;
  if (t < 384 * 960) {                        // WfT[o][k]: [Wout|Woutb|Woutpts]
    int o = t / 960, k = t % 960;
    float v;
    if (k < 192)      v = Wout[k * 384 + o];
    else if (k < 576) v = Woutb[(k - 192) * 384 + o];
    else              v = Woutpts[(k - 576) * 384 + o];
    WfT[o * 960 + k] = (bf16)v;
  }
}

// ---- kernel 2: s layernorm -> bf16 ---------------------------------------

__global__ void k_s_ln(const float* __restrict__ s, const float* __restrict__ g,
                       const float* __restrict__ b, bf16* __restrict__ out) {
  __shared__ float red[128];
  int row = blockIdx.x, t = threadIdx.x;
  const float* sr = s + row * CS;
  float vals[3]; float sum = 0.f;
#pragma unroll
  for (int e = 0; e < 3; ++e) { vals[e] = sr[t + 128 * e]; sum += vals[e]; }
  red[t] = sum; __syncthreads();
  for (int off = 64; off > 0; off >>= 1) { if (t < off) red[t] += red[t + off]; __syncthreads(); }
  float mean = red[0] * (1.f / CS); __syncthreads();
  float sq = 0.f;
#pragma unroll
  for (int e = 0; e < 3; ++e) { float d = vals[e] - mean; sq += d * d; }
  red[t] = sq; __syncthreads();
  for (int off = 64; off > 0; off >>= 1) { if (t < off) red[t] += red[t + off]; __syncthreads(); }
  float rsig = rsqrtf(red[0] * (1.f / CS) + 1e-5f);
#pragma unroll
  for (int e = 0; e < 3; ++e) {
    int c = t + 128 * e;
    out[row * CS + c] = (bf16)((vals[e] - mean) * rsig * g[c] + b[c]);
  }
}

// ---- kernel 3: sproj = s_ln @ Wall (bf16 WMMA) ---------------------------

__global__ void k_sproj(const bf16* __restrict__ A, const bf16* __restrict__ BT,
                        float* __restrict__ D) {
  int lane = threadIdx.x, half = lane >> 4, lm = lane & 15;
  int r0 = blockIdx.x * 16, c0 = blockIdx.y * 16;
  v8f acc = vzero8();
  const bf16* arow = A + (r0 + lm) * CS;
  const bf16* brow = BT + (c0 + lm) * CS;
  for (int k0 = 0; k0 < CS; k0 += 32)
    acc = wmma_bf(a_frag_bf(arow + k0 + half * 16), b_frag_bf(brow + k0 + half * 16), acc);
#pragma unroll
  for (int v = 0; v < 8; ++v)
    D[(r0 + v + 8 * half) * 1152 + c0 + lm] = acc[v];
}

// ---- kernel 4: build augmented Q/K (f16) and combined V^T (bf16) ---------

__global__ void k_aug(const float* __restrict__ sproj, const float* __restrict__ rot,
                      const float* __restrict__ trans, const float* __restrict__ head_w,
                      _Float16* __restrict__ Qaug, _Float16* __restrict__ Kaug,
                      bf16* __restrict__ VcombT) {
  int t = blockIdx.x * blockDim.x + threadIdx.x;
  if (t >= NRES * NH) return;
  int r = t / NH, h = t % NH;
  const float* sp = sproj + r * 1152;
  const float* R = rot + r * 9;
  const float* T = trans + r * 3;
  float hw = log1pf(expf(head_w[h])) * 0.13608276348795434f;  // sqrt(1/54)
  const float s1 = 0.14433756729740643f;                      // sqrt(1/48)
  _Float16* Qa = Qaug + (size_t)(h * NRES + r) * 32;
  _Float16* Ka = Kaug + (size_t)(h * NRES + r) * 32;
#pragma unroll
  for (int c = 0; c < 16; ++c) {
    Qa[c] = (_Float16)(sp[h * 16 + c] * s1);
    Ka[c] = (_Float16)(sp[192 + h * 16 + c]);
  }
  float qn = 0.f, kn = 0.f;
#pragma unroll
  for (int p = 0; p < PQK; ++p) {
    float ql[3], kl[3];
#pragma unroll
    for (int x = 0; x < 3; ++x) {
      ql[x] = sp[576 + (h * PQK + p) * 3 + x];
      kl[x] = sp[720 + (h * PQK + p) * 3 + x];
    }
#pragma unroll
    for (int x = 0; x < 3; ++x) {
      float qg = R[x*3+0]*ql[0] + R[x*3+1]*ql[1] + R[x*3+2]*ql[2] + T[x];
      float kg = R[x*3+0]*kl[0] + R[x*3+1]*kl[1] + R[x*3+2]*kl[2] + T[x];
      qn += qg * qg; kn += kg * kg;
      Qa[16 + p * 3 + x] = (_Float16)(qg * hw);
      Ka[16 + p * 3 + x] = (_Float16)kg;
    }
  }
  Qa[28] = (_Float16)1.0f; Qa[29] = (_Float16)(-0.5f * hw * qn);
  Qa[30] = (_Float16)0.f;  Qa[31] = (_Float16)0.f;
  Ka[28] = (_Float16)(-0.5f * hw * kn); Ka[29] = (_Float16)1.0f;
  Ka[30] = (_Float16)0.f;  Ka[31] = (_Float16)0.f;
#pragma unroll
  for (int c = 0; c < 16; ++c)
    VcombT[(size_t)(h * 48 + c) * NRES + r] = (bf16)sp[384 + h * 16 + c];
#pragma unroll
  for (int p = 0; p < PV; ++p) {
    float vl[3];
#pragma unroll
    for (int x = 0; x < 3; ++x) vl[x] = sp[864 + (h * PV + p) * 3 + x];
#pragma unroll
    for (int x = 0; x < 3; ++x) {
      float vg = R[x*3+0]*vl[0] + R[x*3+1]*vl[1] + R[x*3+2]*vl[2] + T[x];
      VcombT[(size_t)(h * 48 + 16 + p * 3 + x) * NRES + r] = (bf16)vg;
    }
  }
#pragma unroll
  for (int c = 40; c < 48; ++c)
    VcombT[(size_t)(h * 48 + c) * NRES + r] = (bf16)0.f;
}

// ---- kernel 5: fused z layernorm + [Wb|Wdz] projection (the big pass) ----
// Single LDS arena: [0, 48*128) bf16 = WzT copy (async-loaded),
//                   [48*128, ...)   = 8 per-wave 16x128 bf16 A-tiles.

#define ZP_WOFF (48 * CZ)

__global__ void k_zproj(const float* __restrict__ z, const float* __restrict__ gz,
                        const float* __restrict__ bz, const bf16* __restrict__ WzT,
                        float* __restrict__ b_out, bf16* __restrict__ pairgT) {
  __shared__ bf16 lds[ZP_WOFF + 8 * 16 * CZ];

  // async preload of WzT (12 KB) into LDS[0..) : 256 threads x 3 x 16B
  {
    unsigned long long gbase = (unsigned long long)WzT;
#pragma unroll
    for (int c = 0; c < 3; ++c) {
      unsigned loff = (unsigned)((c * 256 + threadIdx.x) * 16);  // LDS byte offset
      unsigned long long ga = gbase + loff;
      asm volatile("global_load_async_to_lds_b128 %0, %1, off"
                   :: "v"(loff), "v"(ga) : "memory");
    }
  }

  // prefetch the next block's z slice (64 KB, one byte per 256B line per thread)
  if (blockIdx.x + 1 < gridDim.x) {
    const char* nxt = (const char*)(z + (size_t)(blockIdx.x + 1) * 128 * CZ)
                      + (size_t)threadIdx.x * 256;
    __builtin_prefetch(nxt, 0, 0);
  }

  int lane = threadIdx.x & 31, wid = threadIdx.x >> 5;
  int half = lane >> 4, lm = lane & 15;
  int p0 = blockIdx.x * 128 + wid * 16;
  int p = p0 + lm;
  const float* zr = z + (size_t)p * CZ + half * 64;
  float x[64]; float sum = 0.f;
#pragma unroll
  for (int t = 0; t < 16; ++t) {
    v4f v = *(const v4f*)(zr + t * 4);
#pragma unroll
    for (int e = 0; e < 4; ++e) { x[t * 4 + e] = v[e]; sum += v[e]; }
  }
  sum += __shfl_xor(sum, 16, 32);
  float mean = sum * (1.f / CZ);
  float sq = 0.f;
#pragma unroll
  for (int t = 0; t < 64; ++t) { float d = x[t] - mean; sq += d * d; }
  sq += __shfl_xor(sq, 16, 32);
  float rsig = rsqrtf(sq * (1.f / CZ) + 1e-5f);

  bf16* lrow = lds + ZP_WOFF + wid * 16 * CZ + lm * CZ + half * 64;
  const float* gzp = gz + half * 64;
  const float* bzp = bz + half * 64;
#pragma unroll
  for (int t4 = 0; t4 < 16; ++t4) {
    v4f gv = *(const v4f*)(gzp + t4 * 4);
    v4f bv = *(const v4f*)(bzp + t4 * 4);
#pragma unroll
    for (int e = 0; e < 4; ++e)
      lrow[t4 * 4 + e] = (bf16)((x[t4 * 4 + e] - mean) * rsig * gv[e] + bv[e]);
  }

  // wait for the async WzT copy, then block barrier covers both LDS regions
  asm volatile("s_wait_asynccnt 0x0" ::: "memory");
  __syncthreads();

  const bf16* arow = lds + ZP_WOFF + wid * 16 * CZ + lm * CZ;
#pragma unroll
  for (int nt = 0; nt < 3; ++nt) {
    v8f acc = vzero8();
    const bf16* brow = lds + (nt * 16 + lm) * CZ;   // WzT from LDS
#pragma unroll
    for (int k0 = 0; k0 < CZ; k0 += 32)
      acc = wmma_bf(a_frag_bf(arow + k0 + half * 16), b_frag_bf(brow + k0 + half * 16), acc);
#pragma unroll
    for (int v = 0; v < 8; ++v) {
      int pp = p0 + v + 8 * half;
      int col = nt * 16 + lm;
      if (col < 12) {
        b_out[(size_t)pp * NH + col] = acc[v];
      } else {
        int c = col - 12;
        if (c < CZ4) {
          int ii = pp / NRES, jj = pp % NRES;
          pairgT[((size_t)ii * CZ4 + c) * NRES + jj] = (bf16)acc[v];
        }
      }
    }
  }
}

// ---- kernel 6: attention core: logits (f16 WMMA) + softmax + o/o_pt -----

__global__ void k_attn(const _Float16* __restrict__ Qaug, const _Float16* __restrict__ Kaug,
                       const float* __restrict__ b_ws, const bf16* __restrict__ VcombT,
                       const int* __restrict__ idx, const float* __restrict__ s_mask,
                       float* __restrict__ logitsbuf, bf16* __restrict__ probs,
                       float* __restrict__ o_ws, float* __restrict__ opt_ws,
                       float* __restrict__ invs) {
  __shared__ bf16 lds[16 * 32];
  int lane = threadIdx.x, half = lane >> 4, lm = lane & 15;
  int tile = blockIdx.x;
  int i0 = (tile / NH) * 16, h = tile % NH;
  const float S2 = 0.57735026919f;  // sqrt(1/3)

  int ii[8]; float mi[8]; float rm[8]; float rs[8];
#pragma unroll
  for (int v = 0; v < 8; ++v) {
    int i = i0 + v + 8 * half;
    ii[v] = idx[i]; mi[v] = s_mask[ii[v]];
    rm[v] = -1e30f; rs[v] = 0.f;
  }
  v16h aq = a_frag_h(Qaug + (size_t)(h * NRES + i0 + lm) * 32 + half * 16);

  // pass 1: logits, store fp32, track row max
  for (int jt = 0; jt < 48; ++jt) {
    int j0 = jt * 16;
    v16h bk = b_frag_h(Kaug + (size_t)(h * NRES + j0 + lm) * 32 + half * 16);
    v8f acc = vzero8();
    acc = wmma_h(aq, bk, acc);
    int j = j0 + lm;
    int jid = idx[j]; float mj = s_mask[jid];
#pragma unroll
    for (int v = 0; v < 8; ++v) {
      float bias = b_ws[((size_t)ii[v] * NRES + jid) * NH + h];
      float l = acc[v] + S2 * (bias + 1e5f * (mi[v] * mj - 1.0f));
      int i = i0 + v + 8 * half;
      logitsbuf[((size_t)i * NH + h) * NRES + j] = l;
      rm[v] = fmaxf(rm[v], l);
    }
  }
#pragma unroll
  for (int m = 1; m < 16; m <<= 1)
#pragma unroll
    for (int v = 0; v < 8; ++v) rm[v] = fmaxf(rm[v], __shfl_xor(rm[v], m, 32));

  // pass 2: exp, probs (unnormalized bf16), fused o / o_pt GEMM
  v8f acc_o[3]; acc_o[0] = vzero8(); acc_o[1] = vzero8(); acc_o[2] = vzero8();
  for (int kt = 0; kt < 24; ++kt) {
    int j0 = kt * 32;
#pragma unroll
    for (int js = 0; js < 2; ++js) {
      int j = j0 + js * 16 + lm;
#pragma unroll
      for (int v = 0; v < 8; ++v) {
        int i = i0 + v + 8 * half;
        float l = logitsbuf[((size_t)i * NH + h) * NRES + j];
        float e = __expf(l - rm[v]);
        rs[v] += e;
        bf16 eb = (bf16)e;
        lds[(v + 8 * half) * 32 + js * 16 + lm] = eb;
        probs[((size_t)i * 16 + h) * NRES + j] = eb;
      }
    }
    __syncthreads();
    v16bf af = a_frag_bf(lds + lm * 32 + half * 16);
#pragma unroll
    for (int nt = 0; nt < 3; ++nt)
      acc_o[nt] = wmma_bf(af,
        b_frag_bf(VcombT + (size_t)(h * 48 + nt * 16 + lm) * NRES + j0 + half * 16),
        acc_o[nt]);
    __syncthreads();
  }
#pragma unroll
  for (int m = 1; m < 16; m <<= 1)
#pragma unroll
    for (int v = 0; v < 8; ++v) rs[v] += __shfl_xor(rs[v], m, 32);
#pragma unroll
  for (int v = 0; v < 8; ++v) {
    int i = i0 + v + 8 * half;
    float inv = 1.f / rs[v];
    o_ws[((size_t)i * NH + h) * CH + lm] = acc_o[0][v] * inv;
    opt_ws[((size_t)i * NH + h) * 24 + lm] = acc_o[1][v] * inv;
    if (lm < 8) opt_ws[((size_t)i * NH + h) * 24 + 16 + lm] = acc_o[2][v] * inv;
    if (lm == 0) invs[(size_t)i * NH + h] = inv;
  }
}

// ---- kernel 7: o_pair = probs @ pair_g  (per-query GEMM, bf16 WMMA) ------

__global__ void k_opair(const bf16* __restrict__ probs, const bf16* __restrict__ pairgT,
                        const int* __restrict__ idx, const float* __restrict__ invs,
                        float* __restrict__ opair_ws) {
  __shared__ bf16 lds[32 * 32];
  int lane = threadIdx.x, half = lane >> 4, lm = lane & 15;
  int i = blockIdx.x;
  int irow = idx[i];
  v8f acc[2]; acc[0] = vzero8(); acc[1] = vzero8();
  for (int kt = 0; kt < 24; ++kt) {
    int j0 = kt * 32;
    int jj = idx[j0 + lane];
    const bf16* src = pairgT + (size_t)irow * CZ4 * NRES + jj;
#pragma unroll
    for (int c = 0; c < 32; ++c) lds[c * 32 + lane] = src[(size_t)c * NRES];
    __syncthreads();
    v16bf af = a_frag_bf(probs + ((size_t)i * 16 + lm) * NRES + j0 + half * 16);
#pragma unroll
    for (int nt = 0; nt < 2; ++nt)
      acc[nt] = wmma_bf(af, b_frag_bf(lds + (nt * 16 + lm) * 32 + half * 16), acc[nt]);
    __syncthreads();
  }
#pragma unroll
  for (int nt = 0; nt < 2; ++nt)
#pragma unroll
    for (int v = 0; v < 8; ++v) {
      int h = v + 8 * half;
      if (h < NH) {
        float inv = invs[(size_t)i * NH + h];
        opair_ws[((size_t)i * NH + h) * CZ4 + nt * 16 + lm] = acc[nt][v] * inv;
      }
    }
}

// ---- kernel 8: build final feature vector (invert-apply + norms) ---------

__global__ void k_feat(const float* __restrict__ o_ws, const float* __restrict__ opt_ws,
                       const float* __restrict__ opair_ws, const float* __restrict__ rot,
                       const float* __restrict__ trans, bf16* __restrict__ F) {
  int t = blockIdx.x * blockDim.x + threadIdx.x;
  if (t >= NRES * NH) return;
  int i = t / NH, h = t % NH;
  bf16* f = F + (size_t)i * 960;
#pragma unroll
  for (int c = 0; c < 16; ++c) f[h * 16 + c] = (bf16)o_ws[((size_t)i * NH + h) * CH + c];
#pragma unroll
  for (int c = 0; c < 32; ++c) f[192 + h * 32 + c] = (bf16)opair_ws[((size_t)i * NH + h) * CZ4 + c];
  const float* R = rot + i * 9;
  const float* T = trans + i * 3;
#pragma unroll
  for (int p = 0; p < PV; ++p) {
    float g[3];
#pragma unroll
    for (int x = 0; x < 3; ++x) g[x] = opt_ws[((size_t)i * NH + h) * 24 + p * 3 + x] - T[x];
    float n2 = 0.f;
#pragma unroll
    for (int x = 0; x < 3; ++x) {
      float lx = R[0 * 3 + x] * g[0] + R[1 * 3 + x] * g[1] + R[2 * 3 + x] * g[2];  // R^T
      f[576 + (h * PV + p) * 3 + x] = (bf16)lx;
      n2 += lx * lx;
    }
    f[864 + h * PV + p] = (bf16)sqrtf(n2 + 1e-8f);
  }
}

// ---- kernel 9: out = F @ Wf  (bf16 WMMA) ---------------------------------

__global__ void k_final(const bf16* __restrict__ F, const bf16* __restrict__ WfT,
                        float* __restrict__ out) {
  int lane = threadIdx.x, half = lane >> 4, lm = lane & 15;
  int r0 = blockIdx.x * 16, c0 = blockIdx.y * 16;
  v8f acc = vzero8();
  const bf16* arow = F + (size_t)(r0 + lm) * 960;
  const bf16* brow = WfT + (size_t)(c0 + lm) * 960;
  for (int k0 = 0; k0 < 960; k0 += 32)
    acc = wmma_bf(a_frag_bf(arow + k0 + half * 16), b_frag_bf(brow + k0 + half * 16), acc);
#pragma unroll
  for (int v = 0; v < 8; ++v)
    out[(size_t)(r0 + v + 8 * half) * CS + c0 + lm] = acc[v];
}

// ---- launcher ------------------------------------------------------------

extern "C" void kernel_launch(void* const* d_in, const int* in_sizes, int n_in,
                              void* d_out, int out_size, void* d_ws, size_t ws_size,
                              hipStream_t stream) {
  const float* s       = (const float*)d_in[0];
  const float* z       = (const float*)d_in[1];
  const float* rot     = (const float*)d_in[2];
  const float* trans   = (const float*)d_in[3];
  const float* s_mask  = (const float*)d_in[4];
  const float* g_s     = (const float*)d_in[5];
  const float* b_s     = (const float*)d_in[6];
  const float* g_z     = (const float*)d_in[7];
  const float* b_z     = (const float*)d_in[8];
  const float* Wq      = (const float*)d_in[9];
  const float* Wk      = (const float*)d_in[10];
  const float* Wv      = (const float*)d_in[11];
  const float* Wqp     = (const float*)d_in[12];
  const float* Wkp     = (const float*)d_in[13];
  const float* Wvp     = (const float*)d_in[14];
  const float* Wb      = (const float*)d_in[15];
  const float* Wdz     = (const float*)d_in[16];
  const float* head_w  = (const float*)d_in[17];
  const float* Wout    = (const float*)d_in[18];
  const float* Woutb   = (const float*)d_in[19];
  const float* Woutpts = (const float*)d_in[20];
  const int*   idx     = (const int*)d_in[21];
  float* out = (float*)d_out;

  char* base = (char*)d_ws;
  size_t off = 0;
  auto alloc = [&](size_t bytes) -> void* {
    void* p = base + off;
    off = (off + bytes + 255) & ~(size_t)255;
    return p;
  };
  bf16*     s_bf      = (bf16*)alloc((size_t)NRES * CS * 2);
  bf16*     WallT     = (bf16*)alloc((size_t)1152 * 384 * 2);
  bf16*     WzT       = (bf16*)alloc((size_t)48 * 128 * 2);
  bf16*     WfT       = (bf16*)alloc((size_t)384 * 960 * 2);
  float*    sproj     = (float*)alloc((size_t)NRES * 1152 * 4);
  _Float16* Qaug      = (_Float16*)alloc((size_t)NH * NRES * 32 * 2);
  _Float16* Kaug      = (_Float16*)alloc((size_t)NH * NRES * 32 * 2);
  bf16*     VcombT    = (bf16*)alloc((size_t)NH * 48 * NRES * 2);
  float*    b_ws      = (float*)alloc((size_t)NRES * NRES * NH * 4);
  bf16*     pairgT    = (bf16*)alloc((size_t)NRES * CZ4 * NRES * 2);
  float*    logitsbuf = (float*)alloc((size_t)NRES * NH * NRES * 4);
  bf16*     probs     = (bf16*)alloc((size_t)NRES * 16 * NRES * 2);
  float*    o_ws      = (float*)alloc((size_t)NRES * NH * CH * 4);
  float*    opt_ws    = (float*)alloc((size_t)NRES * NH * 24 * 4);
  float*    invs      = (float*)alloc((size_t)NRES * NH * 4);
  float*    opair_ws  = (float*)alloc((size_t)NRES * NH * CZ4 * 4);
  bf16*     F         = (bf16*)alloc((size_t)NRES * 960 * 2);
  if (off > ws_size) return;  // workspace too small; bail deterministically

  int packN = 1152 * 384 + 48 * 128 + 384 * 960;
  k_pack<<<(packN + 255) / 256, 256, 0, stream>>>(Wq, Wk, Wv, Wqp, Wkp, Wvp, Wb, Wdz,
                                                  Wout, Woutb, Woutpts, WallT, WzT, WfT);
  k_s_ln<<<NRES, 128, 0, stream>>>(s, g_s, b_s, s_bf);
  k_sproj<<<dim3(NRES / 16, 1152 / 16), 32, 0, stream>>>(s_bf, WallT, sproj);
  k_aug<<<(NRES * NH + 255) / 256, 256, 0, stream>>>(sproj, rot, trans, head_w,
                                                     Qaug, Kaug, VcombT);
  k_zproj<<<(NRES * NRES) / 128, 256, 0, stream>>>(z, g_z, b_z, WzT, b_ws, pairgT);
  k_attn<<<(NRES / 16) * NH, 32, 0, stream>>>(Qaug, Kaug, b_ws, VcombT, idx, s_mask,
                                              logitsbuf, probs, o_ws, opt_ws, invs);
  k_opair<<<NRES, 32, 0, stream>>>(probs, pairgT, idx, invs, opair_ws);
  k_feat<<<(NRES * NH + 255) / 256, 256, 0, stream>>>(o_ws, opt_ws, opair_ws, rot, trans, F);
  k_final<<<dim3(NRES / 16, CS / 16), 32, 0, stream>>>(F, WfT, out);
}
